// HeteroGNNLayer_10402410791052
// MI455X (gfx1250) — matrix operand order repaired and checked by reading, not compiled, not run on hardware
//
#include <hip/hip_runtime.h>

#define D_ 256
#define H_ 8
#define DH_ 32
#define NI_ 40000
#define NU_ 20000
#define NF_ 30000
#define NP_ 20000

typedef __bf16 bf16;
typedef __attribute__((ext_vector_type(16))) __bf16 v16bf;
typedef __attribute__((ext_vector_type(8)))  float  v8f;

// ---------------- utility fills ----------------
__global__ __launch_bounds__(256) void fill_f32_k(float* p, float v, int n) {
    int i = blockIdx.x * 256 + threadIdx.x;
    if (i < n) p[i] = v;
}
__global__ __launch_bounds__(256) void fill_u32_k(unsigned* p, unsigned v, int n) {
    int i = blockIdx.x * 256 + threadIdx.x;
    if (i < n) p[i] = v;
}

// ---------------- weight pack: Wb[n][k] = bf16(W[k][n]) ----------------
__global__ __launch_bounds__(256) void pack_wT_k(const float* __restrict__ W, bf16* __restrict__ Wb) {
    int t = blockIdx.x * 256 + threadIdx.x;   // 65536 threads
    int n = t >> 8, k = t & 255;
    Wb[n * D_ + k] = (bf16)W[k * D_ + n];
}

// ---------------- WMMA GEMM: C[N,256] (+)= A[N,256] @ W + bias ----------------
// One wave computes a 16x64 output strip (4 column tiles): A fragment is loaded/
// converted once per K-step and reused against 4 B fragments -> 32 WMMAs/wave.
__global__ __launch_bounds__(256) void gemm_xW_k(const float* __restrict__ A,
                                                 const bf16* __restrict__ Wb,
                                                 float* __restrict__ C,
                                                 const float* __restrict__ bias,
                                                 int numRowTiles, int accumulate) {
    int wave = blockIdx.x * 8 + (threadIdx.x >> 5);
    int lane = threadIdx.x & 31;
    int numStrips = numRowTiles * 4;           // 4 strips of 64 cols (256 total)
    if (wave >= numStrips) return;             // wave-uniform: EXEC stays all-ones
    int rt = wave >> 2, cg = wave & 3;
    int m0 = rt * 16, n0 = cg * 64;
    int lrow = lane & 15;
    int hi   = lane >> 4;

    v8f acc[4];
#pragma unroll
    for (int j = 0; j < 4; ++j) acc[j] = (v8f){0.f, 0.f, 0.f, 0.f, 0.f, 0.f, 0.f, 0.f};

    const float* arow = A + (size_t)(m0 + lrow) * D_;

#pragma unroll
    for (int k0 = 0; k0 < D_; k0 += 32) {
        // A fragment (contiguous K order): elems 0..15 = K ka..ka+15, ka = k0 + 16*(lane/16)
        int ka = k0 + (hi ? 16 : 0);
        float4 f0 = *(const float4*)(arow + ka);
        float4 f1 = *(const float4*)(arow + ka + 4);
        float4 f2 = *(const float4*)(arow + ka + 8);
        float4 f3 = *(const float4*)(arow + ka + 12);
        v16bf a;
        a[0]  = (bf16)f0.x; a[1]  = (bf16)f0.y; a[2]  = (bf16)f0.z; a[3]  = (bf16)f0.w;
        a[4]  = (bf16)f1.x; a[5]  = (bf16)f1.y; a[6]  = (bf16)f1.z; a[7]  = (bf16)f1.w;
        a[8]  = (bf16)f2.x; a[9]  = (bf16)f2.y; a[10] = (bf16)f2.z; a[11] = (bf16)f2.w;
        a[12] = (bf16)f3.x; a[13] = (bf16)f3.y; a[14] = (bf16)f3.z; a[15] = (bf16)f3.w;
        // B fragments (32x16): lanes 0-15 hold K k0..k0+15, lanes 16-31 K k0+16..k0+31
        int kb = k0 + (hi ? 16 : 0);
#pragma unroll
        for (int j = 0; j < 4; ++j) {
            v16bf b = *(const v16bf*)(Wb + (size_t)(n0 + 16 * j + lrow) * D_ + kb);
            acc[j] = __builtin_amdgcn_wmma_f32_16x16x32_bf16(false, a, false, b, (short)0,
                                                             acc[j], false, false);
        }
    }

    // D layout: VGPR r -> row m0 + r (+8 for lanes 16-31), col tile base + lane%16
    int rbase = m0 + (hi ? 8 : 0);
#pragma unroll
    for (int j = 0; j < 4; ++j) {
        int col = n0 + 16 * j + lrow;
        float bv = bias ? bias[col] : 0.0f;
#pragma unroll
        for (int r = 0; r < 8; ++r) {
            float v = acc[j][r] + bv;
            float* p = C + (size_t)(rbase + r) * D_ + col;
            if (accumulate) *p += v; else *p = v;
        }
    }
}

// ---------------- SAGE mean aggregation ----------------
__global__ __launch_bounds__(256) void scatter_mean_k(const float* __restrict__ xs,
                                                      const int* __restrict__ src,
                                                      const int* __restrict__ dst,
                                                      float* tmp, float* cnt, int E) {
    int t = blockIdx.x * 256 + threadIdx.x;    // E*64 threads, 4 floats each
    int e = t >> 6, j = t & 63;
    if (e >= E) return;
    int s = src[e], d = dst[e];
    float4 v = *(const float4*)(xs + (size_t)s * D_ + j * 4);
    float* o = tmp + (size_t)d * D_ + j * 4;
    atomicAdd(o + 0, v.x); atomicAdd(o + 1, v.y);
    atomicAdd(o + 2, v.z); atomicAdd(o + 3, v.w);
    if (j == 0) atomicAdd(cnt + d, 1.0f);
}

__global__ __launch_bounds__(256) void div_mean_k(float* tmp, const float* cnt, int n) {
    int i = blockIdx.x * 256 + threadIdx.x;
    if (i < n) tmp[i] /= fmaxf(cnt[i >> 8], 1.0f);
}

// ---------------- GAT ----------------
__device__ __forceinline__ unsigned encf(float f) {
    unsigned u = __float_as_uint(f);
    return (u & 0x80000000u) ? ~u : (u | 0x80000000u);
}
__device__ __forceinline__ float decf(unsigned u) {
    return __uint_as_float((u & 0x80000000u) ? (u & 0x7fffffffu) : ~u);
}
__device__ __forceinline__ float lrelu(float a) { return a > 0.f ? a : 0.2f * a; }

__global__ __launch_bounds__(256) void gat_logits_k(const float* __restrict__ xp,
                                                    const float* __restrict__ as,
                                                    const float* __restrict__ ad,
                                                    float* al_s, float* al_d, int NH) {
    int t = blockIdx.x * 256 + threadIdx.x;    // N*H threads
    if (t >= NH) return;
    int n = t >> 3, h = t & 7;
    const float* p  = xp + (size_t)n * D_ + h * DH_;
    const float* ws = as + h * DH_;
    const float* wd = ad + h * DH_;
    float ss = 0.f, sd = 0.f;
#pragma unroll
    for (int i = 0; i < DH_; ++i) { float v = p[i]; ss += v * ws[i]; sd += v * wd[i]; }
    al_s[t] = ss; al_d[t] = sd;
}

__global__ __launch_bounds__(256) void gat_max_k(const float* __restrict__ al_s,
                                                 const float* __restrict__ al_d,
                                                 const int* __restrict__ src,
                                                 const int* __restrict__ dst,
                                                 unsigned* mseg, int EH) {
    int t = blockIdx.x * 256 + threadIdx.x;
    if (t >= EH) return;
    int e = t >> 3, h = t & 7;
    float a = lrelu(al_s[src[e] * H_ + h] + al_d[dst[e] * H_ + h]);
    atomicMax(mseg + dst[e] * H_ + h, encf(a));
}

__global__ __launch_bounds__(256) void gat_sum_k(const float* __restrict__ al_s,
                                                 const float* __restrict__ al_d,
                                                 const int* __restrict__ src,
                                                 const int* __restrict__ dst,
                                                 const unsigned* __restrict__ mseg,
                                                 float* sseg, int EH) {
    int t = blockIdx.x * 256 + threadIdx.x;
    if (t >= EH) return;
    int e = t >> 3, h = t & 7;
    float a = lrelu(al_s[src[e] * H_ + h] + al_d[dst[e] * H_ + h]);
    atomicAdd(sseg + dst[e] * H_ + h, __expf(a - decf(mseg[dst[e] * H_ + h])));
}

__global__ __launch_bounds__(256) void gat_agg_k(const float* __restrict__ xp,
                                                 const float* __restrict__ al_s,
                                                 const float* __restrict__ al_d,
                                                 const int* __restrict__ src,
                                                 const int* __restrict__ dst,
                                                 const unsigned* __restrict__ mseg,
                                                 const float* __restrict__ sseg,
                                                 float* out) {
    int e = blockIdx.x;                        // one block per edge
    int t = threadIdx.x;                       // 256 = H*DH
    int h = t >> 5;
    int s = src[e], d = dst[e];
    float a = lrelu(al_s[s * H_ + h] + al_d[d * H_ + h]);
    float w = __expf(a - decf(mseg[d * H_ + h])) / sseg[d * H_ + h];
    atomicAdd(out + (size_t)d * D_ + t, xp[(size_t)s * D_ + t] * w);
}

__global__ __launch_bounds__(256) void add_bias2_k(float* o, const float* b1,
                                                   const float* b2, int n) {
    int i = blockIdx.x * 256 + threadIdx.x;
    if (i < n) { int c = i & 255; o[i] += b1[c] + b2[c]; }
}

// ---------------- SiLU + residual + LayerNorm (in-place on out region) ----------------
__global__ __launch_bounds__(256) void post_ln_k(float* o, const float* __restrict__ x,
                                                 const float* __restrict__ g,
                                                 const float* __restrict__ b) {
    int row = blockIdx.x, c = threadIdx.x;
    size_t idx = (size_t)row * D_ + c;
    float ov = o[idx];
    float h = ov / (1.0f + __expf(-ov)) + x[idx];
    __shared__ float s1[256], s2[256];
    s1[c] = h; s2[c] = h * h;
    __syncthreads();
    for (int off = 128; off > 0; off >>= 1) {
        if (c < off) { s1[c] += s1[c + off]; s2[c] += s2[c + off]; }
        __syncthreads();
    }
    float mu  = s1[0] * (1.0f / 256.0f);
    float var = s2[0] * (1.0f / 256.0f) - mu * mu;
    float r = rsqrtf(var + 1e-5f);
    o[idx] = (h - mu) * r * g[c] + b[c];
}

// ---------------- host orchestration ----------------
extern "C" void kernel_launch(void* const* d_in, const int* in_sizes, int n_in,
                              void* d_out, int out_size, void* d_ws, size_t ws_size,
                              hipStream_t stream) {
    const float* x_ip   = (const float*)d_in[0];
    const float* x_user = (const float*)d_in[1];
    const float* x_file = (const float*)d_in[2];
    const float* x_proc = (const float*)d_in[3];
    const int *e1s = (const int*)d_in[4],  *e1d = (const int*)d_in[5];
    const int *e2s = (const int*)d_in[6],  *e2d = (const int*)d_in[7];
    const int *e3s = (const int*)d_in[8],  *e3d = (const int*)d_in[9];
    const int *e4s = (const int*)d_in[10], *e4d = (const int*)d_in[11];
    const int *e5s = (const int*)d_in[12], *e5d = (const int*)d_in[13];
    const int *e6s = (const int*)d_in[14], *e6d = (const int*)d_in[15];
    const int *e7s = (const int*)d_in[16], *e7d = (const int*)d_in[17];
    int E1 = in_sizes[4],  E2 = in_sizes[6],  E3 = in_sizes[8],  E4 = in_sizes[10];
    int E5 = in_sizes[12], E6 = in_sizes[14], E7 = in_sizes[16];
    const float* g1W = (const float*)d_in[18]; const float* g1as = (const float*)d_in[19];
    const float* g1ad = (const float*)d_in[20]; const float* g1b = (const float*)d_in[21];
    const float* g2W = (const float*)d_in[22]; const float* g2as = (const float*)d_in[23];
    const float* g2ad = (const float*)d_in[24]; const float* g2b = (const float*)d_in[25];
    const float* s1Wl = (const float*)d_in[26]; const float* s1bl = (const float*)d_in[27];
    const float* s1Wr = (const float*)d_in[28];
    const float* s2Wl = (const float*)d_in[29]; const float* s2bl = (const float*)d_in[30];
    const float* s2Wr = (const float*)d_in[31];
    const float* s3Wl = (const float*)d_in[32]; const float* s3bl = (const float*)d_in[33];
    const float* s3Wr = (const float*)d_in[34];
    const float* s4Wl = (const float*)d_in[35]; const float* s4bl = (const float*)d_in[36];
    const float* s4Wr = (const float*)d_in[37];
    const float* s5Wl = (const float*)d_in[38]; const float* s5bl = (const float*)d_in[39];
    const float* s5Wr = (const float*)d_in[40];
    const float* lng_ip = (const float*)d_in[41],  *lnb_ip = (const float*)d_in[42];
    const float* lng_us = (const float*)d_in[43],  *lnb_us = (const float*)d_in[44];
    const float* lng_fi = (const float*)d_in[45],  *lnb_fi = (const float*)d_in[46];
    const float* lng_pr = (const float*)d_in[47],  *lnb_pr = (const float*)d_in[48];

    float* out_ip   = (float*)d_out;
    float* out_user = out_ip   + (size_t)NI_ * D_;
    float* out_file = out_user + (size_t)NU_ * D_;
    float* out_proc = out_file + (size_t)NF_ * D_;

    // workspace layout (all offsets multiples of 32B)
    float* tmp     = (float*)d_ws;                       // NI*256 (max rows)
    float* cnt     = tmp + (size_t)NI_ * D_;             // 40960 (padded)
    float* al_s    = cnt + 40960;                        // NI*H
    float* al_d    = al_s + (size_t)NI_ * H_;
    unsigned* mseg = (unsigned*)(al_d + (size_t)NI_ * H_);
    float* sseg    = (float*)(mseg + (size_t)NI_ * H_);
    bf16* Wb       = (bf16*)(sseg + (size_t)NI_ * H_);   // 256*256 bf16 (reused)

    auto fillF = [&](float* p, float v, int n) {
        fill_f32_k<<<(n + 255) / 256, 256, 0, stream>>>(p, v, n);
    };
    auto fillU = [&](unsigned* p, unsigned v, int n) {
        fill_u32_k<<<(n + 255) / 256, 256, 0, stream>>>(p, v, n);
    };
    auto gemm = [&](const float* A, const float* W, float* C, const float* bias,
                    int N, int acc) {
        pack_wT_k<<<(D_ * D_) / 256, 256, 0, stream>>>(W, Wb);
        int strips = (N / 16) * 4;             // 16x64 strips per wave
        gemm_xW_k<<<(strips + 7) / 8, 256, 0, stream>>>(A, Wb, C, bias, N / 16, acc);
    };
    auto sage = [&](const float* xs, const float* xd, const int* src, const int* dst,
                    int E, const float* Wl, const float* bl, const float* Wr,
                    float* outR, int Nd) {
        fillF(tmp, 0.f, Nd * D_);
        fillF(cnt, 0.f, Nd);
        int tE = E * 64;
        scatter_mean_k<<<(tE + 255) / 256, 256, 0, stream>>>(xs, src, dst, tmp, cnt, E);
        div_mean_k<<<(Nd * D_ + 255) / 256, 256, 0, stream>>>(tmp, cnt, Nd * D_);
        gemm(tmp, Wl, outR, bl, Nd, 1);
        gemm(xd, Wr, outR, nullptr, Nd, 1);
    };
    auto gat = [&](const float* W, const float* as, const float* ad,
                   const int* src, const int* dst, int E) {
        gemm(x_ip, W, tmp, nullptr, NI_, 0);             // xp = x_ip @ W
        gat_logits_k<<<(NI_ * H_ + 255) / 256, 256, 0, stream>>>(tmp, as, ad, al_s, al_d,
                                                                 NI_ * H_);
        fillU(mseg, 0u, NI_ * H_);                        // below enc(-inf)
        fillF(sseg, 0.f, NI_ * H_);
        int eh = E * H_;
        gat_max_k<<<(eh + 255) / 256, 256, 0, stream>>>(al_s, al_d, src, dst, mseg, eh);
        gat_sum_k<<<(eh + 255) / 256, 256, 0, stream>>>(al_s, al_d, src, dst, mseg, sseg, eh);
        gat_agg_k<<<E, 256, 0, stream>>>(tmp, al_s, al_d, src, dst, mseg, sseg, out_ip);
    };

    // zero all output accumulators
    fillF((float*)d_out, 0.f, out_size);

    // ip destination: GAT(e1) + GAT(e2) + SAGE5(user->ip, e7)
    gat(g1W, g1as, g1ad, e1s, e1d, E1);
    gat(g2W, g2as, g2ad, e2s, e2d, E2);
    add_bias2_k<<<(NI_ * D_ + 255) / 256, 256, 0, stream>>>(out_ip, g1b, g2b, NI_ * D_);
    sage(x_user, x_ip, e7s, e7d, E7, s5Wl, s5bl, s5Wr, out_ip, NI_);
    // user destination: SAGE3(ip->user, e5)
    sage(x_ip, x_user, e5s, e5d, E5, s3Wl, s3bl, s3Wr, out_user, NU_);
    // file destination: SAGE1(user->file, e3) + SAGE4(proc->file, e6)
    sage(x_user, x_file, e3s, e3d, E3, s1Wl, s1bl, s1Wr, out_file, NF_);
    sage(x_proc, x_file, e6s, e6d, E6, s4Wl, s4bl, s4Wr, out_file, NF_);
    // proc destination: SAGE2(user->proc, e4)
    sage(x_user, x_proc, e4s, e4d, E4, s2Wl, s2bl, s2Wr, out_proc, NP_);

    // SiLU + residual + LayerNorm per node type (in-place on d_out regions)
    post_ln_k<<<NI_, 256, 0, stream>>>(out_ip,   x_ip,   lng_ip, lnb_ip);
    post_ln_k<<<NU_, 256, 0, stream>>>(out_user, x_user, lng_us, lnb_us);
    post_ln_k<<<NF_, 256, 0, stream>>>(out_file, x_file, lng_fi, lnb_fi);
    post_ln_k<<<NP_, 256, 0, stream>>>(out_proc, x_proc, lng_pr, lnb_pr);
}
